// LSA_75909251990221
// MI455X (gfx1250) — matrix operand order repaired and checked by reading, not compiled, not run on hardware
//
#include <hip/hip_runtime.h>
#include <stdint.h>

// Problem constants (match reference)
#define B_     8
#define N_     1024
#define D_     1024
#define H_     16
#define HD_    64
#define INNER_ 1024
#define E3_    3072
#define M_     (B_ * N_)   // 8192 flattened (b, token) rows

typedef __attribute__((ext_vector_type(16))) __bf16 bf16x16;
typedef __attribute__((ext_vector_type(8)))  float  v8f;
typedef __attribute__((ext_vector_type(4)))  unsigned int u32x4;
typedef __attribute__((ext_vector_type(8)))  int    i32x8;
typedef __attribute__((ext_vector_type(4)))  int    i32x4;
typedef __attribute__((ext_vector_type(4)))  float  f32x4;
typedef __attribute__((ext_vector_type(4)))  unsigned short u16x4;

union FragU { bf16x16 v; u32x4 u[2]; };

__device__ __forceinline__ v8f vzero() {
  v8f z = {0.f, 0.f, 0.f, 0.f, 0.f, 0.f, 0.f, 0.f};
  return z;
}

// float -> bf16 bits, round-to-nearest-even
__device__ __forceinline__ unsigned short f2bf(float f) {
  unsigned int u = __float_as_uint(f);
  u += 0x7FFFu + ((u >> 16) & 1u);
  return (unsigned short)(u >> 16);
}

// A fragment: 16x32 bf16, source row-major [M x K] (ld = K stride in elements).
// Lane l: m = l&15, half = l>>4; VGPR v holds K = (v>=4?16:0) + half*8 + (v&3)*2 .. +1
__device__ __forceinline__ bf16x16 load_frag_a(const unsigned short* src, int ld, int lane) {
  const int m = lane & 15, half = lane >> 4;
  const unsigned short* p = src + (size_t)m * ld + half * 8;
  FragU f;
  f.u[0] = *(const u32x4*)(p);
  f.u[1] = *(const u32x4*)(p + 16);
  return f.v;
}

// B fragment: 32x16 bf16 (K x N), source row-major [N x K] (ld = K stride).
// Lane l: n = l&15, kh = l>>4; VGPR v holds K = (v>=4?16:0) + kh*8 + (v&3)*2 .. +1
__device__ __forceinline__ bf16x16 load_frag_b(const unsigned short* src, int ld, int lane) {
  const int n = lane & 15, kh = lane >> 4;
  const unsigned short* p = src + (size_t)n * ld + kh * 8;
  FragU f;
  f.u[0] = *(const u32x4*)(p);
  f.u[1] = *(const u32x4*)(p + 16);
  return f.v;
}

#define WMMA_BF16(a, b, c) \
  __builtin_amdgcn_wmma_f32_16x16x32_bf16(false, (a), false, (b), (short)0, (c), false, false)

// ---------------------------------------------------------------------------
// TDM: stage a [nrows x rowlen] bf16 tile (row stride = gstride elements) from
// global memory into LDS at byte offset lds_off. 2-D descriptor per
// CDNA5 ISA: count=1, type=2 ("image"),
// data_size=1 (2 bytes), tensor dims == tile dims (no OOB clipping needed).
// This toolchain uses the 6-arg builtin: (g0, g1, g2, g3, g2b, cpol).
// Issued by one wave; caller waits TENSORcnt and barriers.
// ---------------------------------------------------------------------------
__device__ __forceinline__ void tdm_load_tile(unsigned lds_off,
                                              const unsigned short* gsrc,
                                              unsigned rowlen, unsigned nrows,
                                              unsigned gstride) {
  const unsigned long long ga = (unsigned long long)(uintptr_t)gsrc;
  u32x4 g0;
  g0[0] = 1u;                                             // count=1, user mode
  g0[1] = lds_off;                                        // lds_addr [63:32]
  g0[2] = (unsigned)ga;                                   // global_addr lo
  g0[3] = (unsigned)((ga >> 32) & 0x01FFFFFFu) | (2u << 30); // addr hi | type=2
  i32x8 g1;
  g1[0] = (int)(1u << 16);                                // data_size=1 (2B)
  g1[1] = (int)((rowlen & 0xFFFFu) << 16);                // tensor_dim0 lo16
  g1[2] = (int)((rowlen >> 16) | ((nrows & 0xFFFFu) << 16));   // dim0 hi | dim1 lo
  g1[3] = (int)((nrows >> 16) | ((rowlen & 0xFFFFu) << 16));   // dim1 hi | tile_dim0
  g1[4] = (int)(nrows & 0xFFFFu);                         // tile_dim1 (tile_dim2=0)
  g1[5] = (int)gstride;                                   // tensor_dim0_stride lo32
  g1[6] = 0;                                              // stride hi / dim1_stride lo
  g1[7] = 0;
  i32x4 zz4 = {0, 0, 0, 0};
  i32x8 zz8 = {0, 0, 0, 0, 0, 0, 0, 0};
  __builtin_amdgcn_tensor_load_to_lds(g0, g1, zz4, zz4, zz8, 0);
}

// LDS byte offset of a __shared__ pointer: LDS aperture flat address keeps the
// LDS offset in addr[31:0].
__device__ __forceinline__ unsigned lds_offset_of(const void* p) {
  return (unsigned)(uintptr_t)p;
}

// ---------------------------------------------------------------------------
// Kernel 1: fp32 -> bf16 conversion of x, w_qkv, w_out (vectorized x4)
// ---------------------------------------------------------------------------
__global__ void lsa_cvt_kernel(const f32x4* __restrict__ x,
                               const f32x4* __restrict__ wq,
                               const f32x4* __restrict__ wo,
                               u16x4* __restrict__ xb,
                               u16x4* __restrict__ wqb,
                               u16x4* __restrict__ wob) {
  const long n1 = (long)M_ * D_ / 4;
  const long n2 = n1 + (long)E3_ * D_ / 4;
  const long total = n2 + (long)D_ * INNER_ / 4;
  long i = (long)blockIdx.x * blockDim.x + threadIdx.x;
  const long stride = (long)gridDim.x * blockDim.x;
  for (; i < total; i += stride) {
    const f32x4* s; u16x4* d; long k;
    if (i < n1)      { s = x;  d = xb;  k = i; }
    else if (i < n2) { s = wq; d = wqb; k = i - n1; }
    else             { s = wo; d = wob; k = i - n2; }
    f32x4 v = s[k];
    u16x4 o;
    o[0] = f2bf(v[0]); o[1] = f2bf(v[1]); o[2] = f2bf(v[2]); o[3] = f2bf(v[3]);
    d[k] = o;
  }
}

// ---------------------------------------------------------------------------
// Kernel 2: QKV projection. qkv[m, e] = sum_d x[m,d] * w_qkv[e,d]
// Block = 8 waves sharing one TDM-staged 16x1024 A tile in LDS; each wave owns
// a 16x64 output slab. Grid: 512 m-tiles x 6 e-sections = 3072 blocks.
// B fragments double-buffered in registers so loads overlap WMMA.
// ---------------------------------------------------------------------------
__global__ void lsa_qkv_gemm_kernel(const unsigned short* __restrict__ xb,
                                    const unsigned short* __restrict__ wqb,
                                    unsigned short* __restrict__ qbuf,
                                    unsigned short* __restrict__ kbuf,
                                    unsigned short* __restrict__ vtbuf) {
  __shared__ __align__(16) unsigned short atile[16 * D_];   // 32 KB

  const int lane = threadIdx.x & 31;
  const int wslot = threadIdx.x >> 5;
  const int mtile = blockIdx.x & 511;     // 512 tiles of 16 rows
  const int esec = blockIdx.x >> 9;       // 6 sections of 512 cols
  const int e0 = esec * 512 + wslot * 64;

  if (wslot == 0) {
    tdm_load_tile(lds_offset_of(atile), xb + (size_t)mtile * 16 * D_, D_, 16, D_);
    __builtin_amdgcn_s_wait_tensorcnt(0);
  }
  __syncthreads();

  v8f acc[4] = {vzero(), vzero(), vzero(), vzero()};
  const unsigned short* wrow = wqb + (size_t)e0 * D_;

  bf16x16 cb[4];
#pragma unroll
  for (int j = 0; j < 4; ++j)
    cb[j] = load_frag_b(wrow + (size_t)(j * 16) * D_, D_, lane);

#pragma unroll 2
  for (int kk = 0; kk < D_; kk += 32) {
    const int kn = (kk + 32) & (D_ - 1);          // wrap prefetch on last iter
    bf16x16 a = load_frag_a(atile + kk, D_, lane);    // ds_load_b128 x2
    bf16x16 nb[4];
#pragma unroll
    for (int j = 0; j < 4; ++j)
      nb[j] = load_frag_b(wrow + (size_t)(j * 16) * D_ + kn, D_, lane);
#pragma unroll
    for (int j = 0; j < 4; ++j)
      acc[j] = WMMA_BF16(a, cb[j], acc[j]);
#pragma unroll
    for (int j = 0; j < 4; ++j)
      cb[j] = nb[j];
  }

  const int n = lane & 15, mh = lane >> 4;
#pragma unroll
  for (int j = 0; j < 4; ++j) {
    const int e = e0 + j * 16 + n;
#pragma unroll
    for (int r = 0; r < 8; ++r) {
      const int mg = mtile * 16 + mh * 8 + r;       // flattened (b, tok)
      const int bb = mg >> 10, tok = mg & 1023;
      const unsigned short val = f2bf(acc[j][r]);
      if (e < 1024) {                                // Q
        const int h = e >> 6, hd = e & 63;
        qbuf[(((size_t)(bb * H_ + h)) * N_ + tok) * HD_ + hd] = val;
      } else if (e < 2048) {                         // K
        const int e2 = e - 1024, h = e2 >> 6, hd = e2 & 63;
        kbuf[(((size_t)(bb * H_ + h)) * N_ + tok) * HD_ + hd] = val;
      } else {                                       // V (stored transposed)
        const int e2 = e - 2048, h = e2 >> 6, hd = e2 & 63;
        vtbuf[(((size_t)(bb * H_ + h)) * HD_ + hd) * N_ + tok] = val;
      }
    }
  }
}

// ---------------------------------------------------------------------------
// Kernel 3: attention with diagonal self-mask, flash-style online softmax.
// Wave = one (b, h, 16-row q-tile). 8*16*64 = 8192 waves.
// K fragments for the next s-chunk and V fragments for the current chunk are
// issued before the softmax VALU block, which hides their latency.
// ---------------------------------------------------------------------------
__global__ void lsa_attn_kernel(const unsigned short* __restrict__ qbuf,
                                const unsigned short* __restrict__ kbuf,
                                const unsigned short* __restrict__ vtbuf,
                                const float* __restrict__ temp,
                                unsigned short* __restrict__ ao) {
  __shared__ __align__(16) unsigned short pst[8][16][32];  // per-wave P staging

  const int lane = threadIdx.x & 31;
  const int wslot = threadIdx.x >> 5;
  const int w = blockIdx.x * 8 + wslot;
  const int qt = w & 63;            // 64 q-tiles per head
  const int h = (w >> 6) & 15;
  const int b = w >> 10;
  const int n = lane & 15, mh = lane >> 4;
  const float scale = __expf(temp[0]);

  const size_t bh = (size_t)(b * H_ + h);
  const unsigned short* qp = qbuf + (bh * N_ + (size_t)qt * 16) * HD_;
  const unsigned short* kp = kbuf + bh * N_ * HD_;
  const unsigned short* vp = vtbuf + bh * HD_ * N_;

  // Q fragments for the whole q-tile (HD=64 -> two K=32 fragments)
  const bf16x16 aq0 = load_frag_a(qp, HD_, lane);
  const bf16x16 aq1 = load_frag_a(qp + 32, HD_, lane);

  v8f O[4] = {vzero(), vzero(), vzero(), vzero()};
  float mrun[8], lrun[8];
#pragma unroll
  for (int r = 0; r < 8; ++r) { mrun[r] = -3.0e38f; lrun[r] = 0.f; }

  // preload K fragments for s-chunk 0
  bf16x16 bk0 = load_frag_b(kp, HD_, lane);
  bf16x16 bk1 = load_frag_b(kp + 32, HD_, lane);
  bf16x16 bk2 = load_frag_b(kp + (size_t)16 * HD_, HD_, lane);
  bf16x16 bk3 = load_frag_b(kp + (size_t)16 * HD_ + 32, HD_, lane);

  for (int s0 = 0; s0 < N_; s0 += 32) {
    // scores: two 16x16 tiles covering columns [s0, s0+32)
    v8f sA = vzero(), sB = vzero();
    sA = WMMA_BF16(aq0, bk0, sA);
    sA = WMMA_BF16(aq1, bk1, sA);
    sB = WMMA_BF16(aq0, bk2, sB);
    sB = WMMA_BF16(aq1, bk3, sB);

    // issue next-chunk K loads and current-chunk V loads now; the softmax
    // VALU block below covers their latency.
    const int sn = (s0 + 32) & (N_ - 1);
    bf16x16 nk0 = load_frag_b(kp + (size_t)sn * HD_, HD_, lane);
    bf16x16 nk1 = load_frag_b(kp + (size_t)sn * HD_ + 32, HD_, lane);
    bf16x16 nk2 = load_frag_b(kp + (size_t)(sn + 16) * HD_, HD_, lane);
    bf16x16 nk3 = load_frag_b(kp + (size_t)(sn + 16) * HD_ + 32, HD_, lane);
    bf16x16 bv0 = load_frag_b(vp + s0, N_, lane);
    bf16x16 bv1 = load_frag_b(vp + (size_t)16 * N_ + s0, N_, lane);
    bf16x16 bv2 = load_frag_b(vp + (size_t)32 * N_ + s0, N_, lane);
    bf16x16 bv3 = load_frag_b(vp + (size_t)48 * N_ + s0, N_, lane);

#pragma unroll
    for (int r = 0; r < 8; ++r) {
      const int m = mh * 8 + r;
      const int qrow = qt * 16 + m;
      float v0 = sA[r] * scale;
      float v1 = sB[r] * scale;
      if (qrow == s0 + n)      v0 = -3.0e38f;   // diagonal self-mask
      if (qrow == s0 + 16 + n) v1 = -3.0e38f;

      float cmax = fmaxf(v0, v1);
#pragma unroll
      for (int off = 1; off < 16; off <<= 1)
        cmax = fmaxf(cmax, __shfl_xor(cmax, off, 32));
      const float mnew = fmaxf(mrun[r], cmax);
      const float corr = __expf(mrun[r] - mnew);
      const float p0 = __expf(v0 - mnew);
      const float p1 = __expf(v1 - mnew);
      float rs = p0 + p1;
#pragma unroll
      for (int off = 1; off < 16; off <<= 1)
        rs += __shfl_xor(rs, off, 32);
      lrun[r] = lrun[r] * corr + rs;
      mrun[r] = mnew;
      O[0][r] *= corr; O[1][r] *= corr; O[2][r] *= corr; O[3][r] *= corr;

      pst[wslot][m][n]      = f2bf(p0);
      pst[wslot][m][16 + n] = f2bf(p1);
    }

    // C-layout -> A-layout via LDS, then P @ V
    FragU pf;
    {
      const int am = lane & 15, ah = lane >> 4;
      const unsigned short* p = &pst[wslot][am][ah * 8];
      pf.u[0] = *(const u32x4*)(p);
      pf.u[1] = *(const u32x4*)(p + 16);
    }
    O[0] = WMMA_BF16(pf.v, bv0, O[0]);
    O[1] = WMMA_BF16(pf.v, bv1, O[1]);
    O[2] = WMMA_BF16(pf.v, bv2, O[2]);
    O[3] = WMMA_BF16(pf.v, bv3, O[3]);

    bk0 = nk0; bk1 = nk1; bk2 = nk2; bk3 = nk3;
  }

  // normalize and write attention output, heads re-interleaved: ao[b, tok, h*64+hd]
#pragma unroll
  for (int r = 0; r < 8; ++r) {
    const int tok = qt * 16 + mh * 8 + r;
    const float inv = 1.0f / lrun[r];
#pragma unroll
    for (int j = 0; j < 4; ++j) {
      ao[((size_t)b * N_ + tok) * INNER_ + h * HD_ + j * 16 + n] = f2bf(O[j][r] * inv);
    }
  }
}

// ---------------------------------------------------------------------------
// Kernel 4: output projection. out[m, d] = sum_e ao[m,e] * w_out[d,e] + b_out[d]
// Same structure as kernel 2: TDM-staged A tile, double-buffered B fragments.
// Grid: 512 m-tiles x 2 d-sections = 1024 blocks.
// ---------------------------------------------------------------------------
__global__ void lsa_out_gemm_kernel(const unsigned short* __restrict__ aob,
                                    const unsigned short* __restrict__ wob,
                                    const float* __restrict__ bias,
                                    float* __restrict__ out) {
  __shared__ __align__(16) unsigned short atile[16 * INNER_];   // 32 KB

  const int lane = threadIdx.x & 31;
  const int wslot = threadIdx.x >> 5;
  const int mtile = blockIdx.x & 511;
  const int d0 = (blockIdx.x >> 9) * 512 + wslot * 64;

  if (wslot == 0) {
    tdm_load_tile(lds_offset_of(atile), aob + (size_t)mtile * 16 * INNER_,
                  INNER_, 16, INNER_);
    __builtin_amdgcn_s_wait_tensorcnt(0);
  }
  __syncthreads();

  v8f acc[4] = {vzero(), vzero(), vzero(), vzero()};
  const unsigned short* wrow = wob + (size_t)d0 * INNER_;

  bf16x16 cb[4];
#pragma unroll
  for (int j = 0; j < 4; ++j)
    cb[j] = load_frag_b(wrow + (size_t)(j * 16) * INNER_, INNER_, lane);

#pragma unroll 2
  for (int kk = 0; kk < INNER_; kk += 32) {
    const int kn = (kk + 32) & (INNER_ - 1);
    bf16x16 a = load_frag_a(atile + kk, INNER_, lane);
    bf16x16 nb[4];
#pragma unroll
    for (int j = 0; j < 4; ++j)
      nb[j] = load_frag_b(wrow + (size_t)(j * 16) * INNER_ + kn, INNER_, lane);
#pragma unroll
    for (int j = 0; j < 4; ++j)
      acc[j] = WMMA_BF16(a, cb[j], acc[j]);
#pragma unroll
    for (int j = 0; j < 4; ++j)
      cb[j] = nb[j];
  }

  const int n = lane & 15, mh = lane >> 4;
#pragma unroll
  for (int j = 0; j < 4; ++j) {
    const int d = d0 + j * 16 + n;
    const float bv = bias[d];
#pragma unroll
    for (int r = 0; r < 8; ++r) {
      const int mg = mtile * 16 + mh * 8 + r;
      out[(size_t)mg * D_ + d] = acc[j][r] + bv;
    }
  }
}

// ---------------------------------------------------------------------------
extern "C" void kernel_launch(void* const* d_in, const int* in_sizes, int n_in,
                              void* d_out, int out_size, void* d_ws, size_t ws_size,
                              hipStream_t stream) {
  const float* x    = (const float*)d_in[0];   // [8,1024,1024]
  const float* wq   = (const float*)d_in[1];   // [3072,1024]
  const float* wo   = (const float*)d_in[2];   // [1024,1024]
  const float* bo   = (const float*)d_in[3];   // [1024]
  const float* temp = (const float*)d_in[4];   // scalar

  const size_t MB = (size_t)1024 * 1024;       // 1M elements
  unsigned short* ws  = (unsigned short*)d_ws;
  unsigned short* xb  = ws;                    // 8M  bf16: x
  unsigned short* wqb = xb + 8 * MB;           // 3M  bf16: w_qkv
  unsigned short* wob = wqb + 3 * MB;          // 1M  bf16: w_out
  unsigned short* qb  = wob + 1 * MB;          // 8M  bf16: Q [b,h,tok,hd]
  unsigned short* kb  = qb + 8 * MB;           // 8M  bf16: K [b,h,tok,hd]
  unsigned short* vtb = kb + 8 * MB;           // 8M  bf16: V^T [b,h,hd,tok]
  unsigned short* aob = vtb + 8 * MB;          // 8M  bf16: attn out [b,tok,inner]
  // total: 44M bf16 elements = 88 MB of workspace

  lsa_cvt_kernel<<<1024, 256, 0, stream>>>(
      (const f32x4*)x, (const f32x4*)wq, (const f32x4*)wo,
      (u16x4*)xb, (u16x4*)wqb, (u16x4*)wob);
  lsa_qkv_gemm_kernel<<<3072, 256, 0, stream>>>(xb, wqb, qb, kb, vtb);
  lsa_attn_kernel<<<1024, 256, 0, stream>>>(qb, kb, vtb, temp, aob);
  lsa_out_gemm_kernel<<<1024, 256, 0, stream>>>(aob, wob, bo, (float*)d_out);
}
